// LiquidOperator_37864431682329
// MI455X (gfx1250) — matrix operand order repaired and checked by reading, not compiled, not run on hardware
//
#include <hip/hip_runtime.h>
#include <hip/hip_bf16.h>

// ---------------------------------------------------------------------------
// LiquidOperator for MI455X (gfx1250, wave32).
//  Phase 1: pre VarEncoder  (5x GEMM 68x68, sigmoid)           -> WMMA f16
//  Phase 2: sensory synapse precompute (parallel over T)       -> VALU/trans
//  Phase 3: sequential LTC scan, 4 independent ODEs, LDS state -> VALU/DS
//  Phase 4: e1 VarEncoder (rows = T*4, dims <= 68)             -> WMMA f16
//  Phase 5: e2 VarEncoder (rows = T, dims <= 272) -> d_out     -> WMMA f16
// ---------------------------------------------------------------------------

#define LEVELS 17
#define VAR_N 4
#define UNITS 51
#define PRED_N 28
#define NCOL (VAR_N * UNITS)   // 204

typedef _Float16 v16h __attribute__((ext_vector_type(16)));
typedef _Float16 v8h  __attribute__((ext_vector_type(8)));
typedef float    v8f  __attribute__((ext_vector_type(8)));

union Frag16 { v16h v; v8h h8[2]; };

__device__ __forceinline__ float fast_sigmoid(float x) {
    // sigmoid(x) = 1 / (1 + exp2(-log2(e) * x)) : v_exp_f32 + v_rcp_f32
    float t = __builtin_amdgcn_exp2f(-1.44269504089f * x);
    return __builtin_amdgcn_rcpf(1.0f + t);
}

// ---------------------------------------------------------------------------
// Generic fused GEMM: Y = act(X[M,K] @ W[N,K]^T + b[N])
// BLOCK = 64 rows x 32 cols, 8 waves, one 16x16 WMMA tile per wave.
// Whole K panel (<= 288) staged in LDS as f16; fragments loaded with the
// documented CDNA5 A(16x32)/B(32x16) f16 lane layouts using b128 DS reads.
// ---------------------------------------------------------------------------
#define GEMM_BM 64
#define GEMM_BN 32
#define GEMM_KMAX 288   // max Kpad over all layers (272 -> 288)

__global__ __launch_bounds__(256)
void mlp_gemm_kernel(const float* __restrict__ X, const float* __restrict__ W,
                     const float* __restrict__ bias, float* __restrict__ Y,
                     int M, int N, int K, int act)
{
    __shared__ _Float16 Alds[GEMM_BM * GEMM_KMAX];
    __shared__ _Float16 Blds[GEMM_BN * GEMM_KMAX];

    const int Kpad = (K + 31) & ~31;
    const int tid  = threadIdx.x;
    const int m0   = blockIdx.x * GEMM_BM;
    const int n0   = blockIdx.y * GEMM_BN;

    // Stage A tile (convert f32 -> f16, zero-pad K and OOB rows)
    for (int idx = tid; idx < GEMM_BM * Kpad; idx += 256) {
        int r = idx / Kpad, k = idx - r * Kpad;
        int gr = m0 + r;
        float v = (gr < M && k < K) ? X[(long)gr * K + k] : 0.0f;
        Alds[r * Kpad + k] = (_Float16)v;
    }
    // Stage B tile: W is (N,K) row-major == exactly the K-major layout we want
    for (int idx = tid; idx < GEMM_BN * Kpad; idx += 256) {
        int r = idx / Kpad, k = idx - r * Kpad;
        int gn = n0 + r;
        float v = (gn < N && k < K) ? W[(long)gn * K + k] : 0.0f;
        Blds[r * Kpad + k] = (_Float16)v;
    }
    __syncthreads();

    const int wave = tid >> 5, lane = tid & 31;
    const int mt = wave & 3, nt = wave >> 2;           // 4 M-tiles x 2 N-tiles
    const int arow  = mt * 16 + (lane & 15);           // A: M = lane%16
    const int brow  = nt * 16 + (lane & 15);           // B: N = lane%16
    const int koffA = (lane >= 16) ? 8 : 0;            // A halves: K {0..7,16..23} / {8..15,24..31}
    const int koffB = (lane >= 16) ? 8 : 0;            // B halves: K {0..7,16..23} / {8..15,24..31}

    v8f acc = {};
    for (int kk = 0; kk < Kpad; kk += 32) {
        Frag16 a, b;
        a.h8[0] = *(const v8h*)&Alds[arow * Kpad + kk + koffA];
        a.h8[1] = *(const v8h*)&Alds[arow * Kpad + kk + koffA + 16];
        b.h8[0] = *(const v8h*)&Blds[brow * Kpad + kk + koffB];
        b.h8[1] = *(const v8h*)&Blds[brow * Kpad + kk + koffB + 16];
        acc = __builtin_amdgcn_wmma_f32_16x16x32_f16(
            /*neg_a=*/false, a.v, /*neg_b=*/false, b.v,
            /*c_mod=*/(short)0, acc, /*reuse_a=*/false, /*reuse_b=*/false);
    }

    // C/D layout: VGPR r holds (M = r + 8*(lane/16), N = lane%16)
    const int col   = n0 + nt * 16 + (lane & 15);
    const int rbase = m0 + mt * 16 + ((lane >> 4) << 3);
    const float bv  = (col < N) ? bias[col] : 0.0f;
#pragma unroll
    for (int r = 0; r < 8; ++r) {
        int row = rbase + r;
        if (row < M && col < N) {
            float vv = acc[r] + bv;
            if (act) vv = fast_sigmoid(vv);
            Y[(long)row * N + col] = vv;
        }
    }
}

// ---------------------------------------------------------------------------
// Phase 2: sensory synapse sums, fully parallel over (t, v, u).
// num_s/den_s[t, v, u] = sum_s w*sig(sigma*(i - mu))*mask * {erev, 1}
// ---------------------------------------------------------------------------
__global__ __launch_bounds__(256)
void sensory_kernel(const float* __restrict__ h,
                    const float* __restrict__ ssigma, const float* __restrict__ smu,
                    const float* __restrict__ sw,     const float* __restrict__ serev,
                    const float* __restrict__ smask,
                    const float* __restrict__ iw,     const float* __restrict__ ib,
                    float* __restrict__ num_s, float* __restrict__ den_s,
                    int T, int Bsz)
{
    long gid = (long)blockIdx.x * blockDim.x + threadIdx.x;
    if (gid >= (long)T * NCOL) return;
    int t   = (int)(gid / NCOL);
    int col = (int)(gid - (long)t * NCOL);
    int v = col / UNITS, u = col - v * UNITS;
    int hrow = (t < Bsz - 1) ? t : (Bsz - 1);          // broadcast of last step
    const float* hp = h + (long)hrow * (VAR_N * LEVELS) + v * LEVELS;

    float num = 0.0f, den = 0.0f;
#pragma unroll
    for (int s = 0; s < LEVELS; ++s) {
        float ival = fmaf(hp[s], iw[v * LEVELS + s], ib[v * LEVELS + s]);
        int p = (v * LEVELS + s) * UNITS + u;
        float a = sw[p] * fast_sigmoid(ssigma[p] * (ival - smu[p])) * smask[p];
        num = fmaf(a, serev[p], num);
        den += a;
    }
    num_s[gid] = num;
    den_s[gid] = den;
}

// ---------------------------------------------------------------------------
// Phase 3: the sequential LTC scan. 4 independent ODEs -> 4 workgroups.
// 512 threads: 51 output columns x 9 presynaptic chunks (6 i's each).
// Synapse params pre-fused into registers; v state + partials in LDS.
// ---------------------------------------------------------------------------
#define SCAN_CH 9
#define SCAN_ISZ 6
#define EPSF 1e-8f

__global__ __launch_bounds__(512)
void ltc_scan_kernel(const float* __restrict__ num_s, const float* __restrict__ den_s,
                     const float* __restrict__ gleak, const float* __restrict__ vleak,
                     const float* __restrict__ cm,
                     const float* __restrict__ sigma, const float* __restrict__ mu,
                     const float* __restrict__ w_rec, const float* __restrict__ erev,
                     const float* __restrict__ mask,
                     const float* __restrict__ out_w, const float* __restrict__ out_b,
                     float* __restrict__ outs, int T)
{
    const int v   = blockIdx.x;          // which of the 4 independent ODEs
    const int tid = threadIdx.x;
    const int j   = tid % UNITS;         // postsynaptic column
    const int c   = tid / UNITS;         // presynaptic chunk
    const bool compute = (tid < UNITS * SCAN_CH);

    __shared__ float vst[UNITS];
    __shared__ float pnum[UNITS * SCAN_CH];
    __shared__ float pden[UNITS * SCAN_CH];

    // Pre-fused synapse params in registers:
    //   exp(-sigma*(v-mu)) = exp2(a2*v + b2)
    float a2[SCAN_ISZ], b2[SCAN_ISZ], wv[SCAN_ISZ], ev[SCAN_ISZ];
    int i0 = c * SCAN_ISZ;
    int ni = 0;
    if (compute) {
        int hi = (i0 + SCAN_ISZ < UNITS) ? (i0 + SCAN_ISZ) : UNITS;
        ni = hi - i0;
        for (int k = 0; k < ni; ++k) {
            int idx = ((v * UNITS) + (i0 + k)) * UNITS + j;
            float sg = sigma[idx];
            a2[k] = -1.44269504089f * sg;
            b2[k] =  1.44269504089f * sg * mu[idx];
            wv[k] = w_rec[idx] * mask[idx];
            ev[k] = erev[idx];
        }
    }

    // Per-column update constants
    float cmt = 0.f, gl = 0.f, glv = 0.f, ow = 0.f, ob = 0.f;
    if (tid < UNITS) {
        cmt = cm[v * UNITS + tid] * 6.0f;    // cm * ODE_UNFOLDS
        gl  = gleak[v * UNITS + tid];
        glv = gl * vleak[v * UNITS + tid];
        if (tid < LEVELS) { ow = out_w[v * LEVELS + tid]; ob = out_b[v * LEVELS + tid]; }
        vst[tid] = 0.0f;                     // v0 = zeros
    }
    __syncthreads();

    for (int t = 0; t < T; ++t) {
        float nsr = 0.f, dsr = 0.f;
        if (tid < UNITS) {
            long base = (long)t * NCOL + v * UNITS + tid;
            nsr = num_s[base];
            dsr = den_s[base];
        }
#pragma unroll 1
        for (int u6 = 0; u6 < 6; ++u6) {
            if (compute) {
                float num = 0.f, den = 0.f;
                for (int k = 0; k < ni; ++k) {
                    float vv  = vst[i0 + k];
                    float ex  = __builtin_amdgcn_exp2f(fmaf(a2[k], vv, b2[k]));
                    float act = wv[k] * __builtin_amdgcn_rcpf(1.0f + ex);
                    num = fmaf(act, ev[k], num);
                    den += act;
                }
                pnum[tid] = num;   // tid == c*51 + j
                pden[tid] = den;
            }
            __syncthreads();
            if (tid < UNITS) {
                float n = nsr, d = dsr;
#pragma unroll
                for (int c2 = 0; c2 < SCAN_CH; ++c2) {
                    n += pnum[c2 * UNITS + tid];
                    d += pden[c2 * UNITS + tid];
                }
                float vold = vst[tid];
                vst[tid] = (cmt * vold + glv + n) / (cmt + gl + d + EPSF);
            }
            __syncthreads();
        }
        if (tid < LEVELS)
            outs[(long)t * (VAR_N * LEVELS) + v * LEVELS + tid] = fmaf(vst[tid], ow, ob);
    }
}

// ---------------------------------------------------------------------------
// Host-side orchestration (graph-capture safe: all launches on `stream`).
// ---------------------------------------------------------------------------
extern "C" void kernel_launch(void* const* d_in, const int* in_sizes, int n_in,
                              void* d_out, int out_size, void* d_ws, size_t ws_size,
                              hipStream_t stream)
{
    const float* x      = (const float*)d_in[0];
    const float* pre_W  = (const float*)d_in[1];
    const float* pre_b  = (const float*)d_in[2];
    const float* gleak  = (const float*)d_in[3];
    const float* vleak  = (const float*)d_in[4];
    const float* cm     = (const float*)d_in[5];
    const float* sigma  = (const float*)d_in[6];
    const float* mu     = (const float*)d_in[7];
    const float* w_rec  = (const float*)d_in[8];
    const float* erev   = (const float*)d_in[9];
    const float* ssig   = (const float*)d_in[10];
    const float* smu    = (const float*)d_in[11];
    const float* sw     = (const float*)d_in[12];
    const float* serev  = (const float*)d_in[13];
    const float* iw     = (const float*)d_in[14];
    const float* ib     = (const float*)d_in[15];
    const float* ow     = (const float*)d_in[16];
    const float* ob     = (const float*)d_in[17];
    const float* mask   = (const float*)d_in[18];
    const float* smask  = (const float*)d_in[19];
    // e1: W0,b0,...,W4,b4 at indices 20..29 ; e2 at 30..39
    const float* e1W[5], *e1b[5], *e2W[5], *e2b[5];
    for (int i = 0; i < 5; ++i) {
        e1W[i] = (const float*)d_in[20 + 2 * i];
        e1b[i] = (const float*)d_in[21 + 2 * i];
        e2W[i] = (const float*)d_in[30 + 2 * i];
        e2b[i] = (const float*)d_in[31 + 2 * i];
    }

    const int Bsz = in_sizes[0] / (VAR_N * LEVELS);     // 32768
    const int T   = Bsz + PRED_N;                       // 32796
    const int M1  = T * VAR_N;                          // 131184 rows for e1

    // Workspace: p0/p1 ping-pong (T*272 each), outs (T*68), num_s/den_s (T*204)
    const long BUF = (long)T * 272;
    float* p0    = (float*)d_ws;
    float* p1    = p0 + BUF;
    float* outs  = p1 + BUF;
    float* num_s = outs + (long)T * (VAR_N * LEVELS);
    float* den_s = num_s + (long)T * NCOL;

    dim3 blk(256);
    auto gemm = [&](const float* in, const float* w, const float* b, float* out,
                    int M, int N, int K, int act) {
        dim3 grid((M + GEMM_BM - 1) / GEMM_BM, (N + GEMM_BN - 1) / GEMM_BN);
        mlp_gemm_kernel<<<grid, blk, 0, stream>>>(in, w, b, out, M, N, K, act);
    };

    // Phase 1: pre encoder (sigmoid x4, linear)
    gemm(x,  pre_W + 0 * 68 * 68, pre_b + 0 * 68, p0, Bsz, 68, 68, 1);
    gemm(p0, pre_W + 1 * 68 * 68, pre_b + 1 * 68, p1, Bsz, 68, 68, 1);
    gemm(p1, pre_W + 2 * 68 * 68, pre_b + 2 * 68, p0, Bsz, 68, 68, 1);
    gemm(p0, pre_W + 3 * 68 * 68, pre_b + 3 * 68, p1, Bsz, 68, 68, 1);
    gemm(p1, pre_W + 4 * 68 * 68, pre_b + 4 * 68, p0, Bsz, 68, 68, 0);  // h = p0

    // Phase 2: sensory sums over all T steps (parallel)
    {
        long total = (long)T * NCOL;
        dim3 grid((unsigned)((total + 255) / 256));
        sensory_kernel<<<grid, blk, 0, stream>>>(p0, ssig, smu, sw, serev, smask,
                                                 iw, ib, num_s, den_s, T, Bsz);
    }

    // Phase 3: sequential scan, one block per independent ODE variable
    ltc_scan_kernel<<<dim3(VAR_N), dim3(512), 0, stream>>>(
        num_s, den_s, gleak, vleak, cm, sigma, mu, w_rec, erev, mask,
        ow, ob, outs, T);

    // Phase 4: e1 encoder on (T*4, 17) rows; outer sigmoid => all layers act=1
    gemm(outs, e1W[0], e1b[0], p1, M1, 17, 17, 1);
    gemm(p1,   e1W[1], e1b[1], p0, M1, 17, 17, 1);
    gemm(p0,   e1W[2], e1b[2], p1, M1, 68, 17, 1);
    gemm(p1,   e1W[3], e1b[3], p0, M1, 68, 68, 1);
    gemm(p0,   e1W[4], e1b[4], p1, M1, 68, 68, 1);   // h1 = p1 viewed as (T, 272)

    // Phase 5: e2 encoder on (T, 272) -> d_out (T, 68)
    gemm(p1, e2W[0], e2b[0], p0, T, 272, 272, 1);
    gemm(p0, e2W[1], e2b[1], p1, T, 272, 272, 1);
    gemm(p1, e2W[2], e2b[2], p0, T, 68, 272, 1);
    gemm(p0, e2W[3], e2b[3], p1, T, 68, 68, 1);
    gemm(p1, e2W[4], e2b[4], (float*)d_out, T, 68, 68, 0);
}
